// GPT2_29248727286444
// MI455X (gfx1250) — matrix operand order repaired and checked by reading, not compiled, not run on hardware
//
#include <hip/hip_runtime.h>
#include <hip/hip_bf16.h>
#include <math.h>

typedef __bf16 bf16_t;
typedef __attribute__((ext_vector_type(16))) __bf16 v16bf;
typedef __attribute__((ext_vector_type(8)))  __bf16 v8bf;
typedef __attribute__((ext_vector_type(8)))  float  v8f;
typedef __attribute__((ext_vector_type(4)))  float  v4f;
typedef __attribute__((ext_vector_type(4)))  int    v4i;
typedef __attribute__((address_space(1))) v4i* gas_v4i;
typedef __attribute__((address_space(3))) v4i* las_v4i;

#define BZ   8
#define SN   1024
#define DM   768
#define HN   12
#define LN_  12
#define DHH  64
#define DFFN 3072
#define VN   50257

#if __has_builtin(__builtin_amdgcn_global_load_async_to_lds_b128) && \
    __has_builtin(__builtin_amdgcn_s_wait_asynccnt)
#define USE_ASYNC 1
#else
#define USE_ASYNC 0
#endif

// ---------------------------------------------------------------------------
// Embedding: x[b,s,:] = tok_emb[ids[b,s]] + pos_emb[s]
// ---------------------------------------------------------------------------
__global__ void gpt2_embed_kernel(const int* __restrict__ ids,
                                  const float* __restrict__ tok,
                                  const float* __restrict__ pos,
                                  float* __restrict__ x, int total) {
  int i = blockIdx.x * blockDim.x + threadIdx.x;
  if (i >= total) return;
  int d = i % DM;
  int t = i / DM;
  int s = t % SN;
  x[i] = tok[(size_t)ids[t] * DM + d] + pos[(size_t)s * DM + d];
}

// ---------------------------------------------------------------------------
// LayerNorm: one block per row; src row = row*rowMul + rowAdd; bf16 out.
// ---------------------------------------------------------------------------
__global__ __launch_bounds__(256)
void gpt2_ln_kernel(const float* __restrict__ x, const float* __restrict__ w,
                    const float* __restrict__ b, bf16_t* __restrict__ out,
                    int rowMul, int rowAdd) {
  int row = blockIdx.x;
  const float* src = x + ((size_t)row * rowMul + rowAdd) * DM;
  int tid = threadIdx.x;
  float vals[3];
  float s1 = 0.f, s2 = 0.f;
#pragma unroll
  for (int i = 0; i < 3; ++i) {
    vals[i] = src[tid + i * 256];
    s1 += vals[i];
    s2 += vals[i] * vals[i];
  }
#pragma unroll
  for (int m = 16; m; m >>= 1) {
    s1 += __shfl_xor(s1, m, 32);
    s2 += __shfl_xor(s2, m, 32);
  }
  __shared__ float r1[8], r2[8];
  int wid = tid >> 5, lane = tid & 31;
  if (lane == 0) { r1[wid] = s1; r2[wid] = s2; }
  __syncthreads();
  if (wid == 0) {
    s1 = (lane < 8) ? r1[lane] : 0.f;
    s2 = (lane < 8) ? r2[lane] : 0.f;
#pragma unroll
    for (int m = 4; m; m >>= 1) {
      s1 += __shfl_xor(s1, m, 32);
      s2 += __shfl_xor(s2, m, 32);
    }
    if (lane == 0) { r1[0] = s1; r2[0] = s2; }
  }
  __syncthreads();
  float mean = r1[0] * (1.f / DM);
  float var  = r2[0] * (1.f / DM) - mean * mean;
  float rstd = rsqrtf(var + 1e-5f);
#pragma unroll
  for (int i = 0; i < 3; ++i) {
    int d = tid + i * 256;
    out[(size_t)row * DM + d] = (bf16_t)((vals[i] - mean) * rstd * w[d] + b[d]);
  }
}

// ---------------------------------------------------------------------------
// WMMA GEMM: C[M,N] = epi(A_bf16[M,K] @ W_f32[K,N | N,K] + bias (+res))
// 256 thr = 8 waves; block tile 128x64; wave tile 16x64 (4 accumulators).
// LDS staged in WMMA fragment layout -> ds_load_b128 frag reads.
// vtrans: write bf16 out as vt[b][h][d][s] (for attention V operand).
// ---------------------------------------------------------------------------
__global__ __launch_bounds__(256)
void gpt2_gemm_kernel(const bf16_t* __restrict__ A, const float* __restrict__ W,
                      const float* __restrict__ bias,
                      const float* __restrict__ residual,
                      float* __restrict__ Cf, bf16_t* __restrict__ Cb,
                      int M, int N, int K, int wt, int gelu, int vtrans) {
  __shared__ __attribute__((aligned(32))) bf16_t AsF[2][8][32][16];
  __shared__ __attribute__((aligned(32))) bf16_t BsF[2][4][32][16];
  const int tid = threadIdx.x;
  const int lane = tid & 31, wid = tid >> 5;
  const int bm = blockIdx.y * 128, bn = blockIdx.x * 64;
  const int mlane = lane & 15, half = lane >> 4;

  const int arow = tid >> 1;
  const int ac0 = (tid & 1) * 8;
  const int amt = arow >> 4;
  const int alane = (arow & 15) + 16 * (tid & 1);
  const int bncol = tid & 63;
  const int bkg = (tid >> 6) * 8;
  const int bnt = bncol >> 4;
  const int blane = (bncol & 15) + 16 * ((bkg >> 3) & 1);
  const int bj0 = (bkg >> 1) & 8;

  v8f acc[4] = {{}, {}, {}, {}};

  auto compute = [&](int bb) {
    v16bf af = *(const v16bf*)&AsF[bb][wid][lane][0];
#pragma unroll
    for (int t = 0; t < 4; ++t) {
      v16bf bfm = *(const v16bf*)&BsF[bb][t][lane][0];
      acc[t] = __builtin_amdgcn_wmma_f32_16x16x32_bf16(
          false, af, false, bfm, (short)0, acc[t], false, false);
    }
  };

  const bool full = (bm + 128 <= M) && (bn + 64 <= N);
  if (full) {
    const bf16_t* aptr = A + (size_t)(bm + arow) * K + ac0;
    auto stageA = [&](int kk, int bb) {
#if USE_ASYNC
      __builtin_amdgcn_global_load_async_to_lds_b128(
          (gas_v4i)(aptr + kk), (las_v4i)&AsF[bb][amt][alane][0], 0, 0);
      __builtin_amdgcn_global_load_async_to_lds_b128(
          (gas_v4i)(aptr + kk + 16), (las_v4i)&AsF[bb][amt][alane][8], 0, 0);
#else
      v8bf a0 = *(const v8bf*)(aptr + kk);
      v8bf a1 = *(const v8bf*)(aptr + kk + 16);
      *(v8bf*)&AsF[bb][amt][alane][0] = a0;
      *(v8bf*)&AsF[bb][amt][alane][8] = a1;
#endif
    };
    auto loadB = [&](int kk, v8bf* breg) {
      if (!wt) {
        const float* wp = W + (size_t)(kk + bkg) * N + bn + bncol;
#pragma unroll
        for (int j = 0; j < 8; ++j) (*breg)[j] = (bf16_t)wp[(size_t)j * N];
      } else {
        const float* wp = W + (size_t)(bn + bncol) * K + kk + bkg;
        v4f w0 = *(const v4f*)wp;
        v4f w1 = *(const v4f*)(wp + 4);
#pragma unroll
        for (int j = 0; j < 4; ++j) {
          (*breg)[j]     = (bf16_t)w0[j];
          (*breg)[j + 4] = (bf16_t)w1[j];
        }
      }
    };
    auto storeB = [&](int bb, v8bf breg) {
      *(v8bf*)&BsF[bb][bnt][blane][bj0] = breg;
    };

    v8bf breg;
    stageA(0, 0);
    loadB(0, &breg);
    storeB(0, breg);
#if USE_ASYNC
    __builtin_amdgcn_s_wait_asynccnt(0);
#endif
    __syncthreads();
    int buf = 0;
    for (int kk = 0; kk < K; kk += 32) {
      const bool more = (kk + 32) < K;
      if (more) {
        stageA(kk + 32, buf ^ 1);
        loadB(kk + 32, &breg);
      }
      compute(buf);
      if (more) {
        storeB(buf ^ 1, breg);
#if USE_ASYNC
        __builtin_amdgcn_s_wait_asynccnt(0);
#endif
      }
      __syncthreads();
      buf ^= 1;
    }
  } else {
    // edge path (lm_head): branchless clamped addresses + value select
    const int gm = bm + arow;
    const int gmc = (gm < M) ? gm : (M - 1);
    const bool mok = (gm < M);
    const int gn = bn + bncol;
    const int gnc = (gn < N) ? gn : (N - 1);
    const bool nok = (gn < N);
    const v8bf zz = {};
    for (int kk = 0; kk < K; kk += 32) {
      v8bf a0 = *(const v8bf*)(A + (size_t)gmc * K + kk + ac0);
      v8bf a1 = *(const v8bf*)(A + (size_t)gmc * K + kk + ac0 + 16);
      *(v8bf*)&AsF[0][amt][alane][0] = mok ? a0 : zz;
      *(v8bf*)&AsF[0][amt][alane][8] = mok ? a1 : zz;
      v8bf breg;
      if (wt) {
        const float* wp = W + (size_t)gnc * K + kk + bkg;
        v4f w0 = *(const v4f*)wp;
        v4f w1 = *(const v4f*)(wp + 4);
#pragma unroll
        for (int j = 0; j < 4; ++j) {
          (breg)[j]     = (bf16_t)w0[j];
          (breg)[j + 4] = (bf16_t)w1[j];
        }
      } else {
        const float* wp = W + (size_t)(kk + bkg) * N + gnc;
#pragma unroll
        for (int j = 0; j < 8; ++j) (breg)[j] = (bf16_t)wp[(size_t)j * N];
      }
      *(v8bf*)&BsF[0][bnt][blane][bj0] = nok ? breg : zz;
      __syncthreads();
      compute(0);
      __syncthreads();
    }
  }

#pragma unroll
  for (int t = 0; t < 4; ++t) {
    int colg = bn + t * 16 + mlane;
    if (colg >= N) continue;
    float bv = bias ? bias[colg] : 0.f;
#pragma unroll
    for (int r = 0; r < 8; ++r) {
      int rowg = bm + wid * 16 + half * 8 + r;
      if (rowg >= M) continue;
      float v = acc[t][r] + bv;
      if (residual) v += residual[(size_t)rowg * N + colg];
      if (gelu) v = 0.5f * v * (1.f + erff(v * 0.70710678118f));
      if (vtrans) {
        // vt[b][h][d][s] : s-contiguous for attention PV vector loads
        int hh = colg >> 6, dd = colg & 63;
        int bb2 = rowg >> 10, ss = rowg & 1023;
        Cb[(((size_t)bb2 * HN + hh) * DHH + dd) * SN + ss] = (bf16_t)v;
      } else {
        size_t o = (size_t)rowg * N + colg;
        if (Cf) Cf[o] = v;
        if (Cb) Cb[o] = (bf16_t)v;
      }
    }
  }
}

// ---------------------------------------------------------------------------
// Flash attention: 1 wave per (b, h, 16-query tile); causal, online softmax.
// Processes k-tiles in PAIRS so the PV WMMA uses its full K=32.
// V is pre-transposed (vt[b][h][d][s]) -> B-frags are 32B vector loads.
// ---------------------------------------------------------------------------
__global__ __launch_bounds__(128)
void gpt2_attn_kernel(const bf16_t* __restrict__ Q, const bf16_t* __restrict__ K,
                      const bf16_t* __restrict__ Vt, bf16_t* __restrict__ O) {
  __shared__ __attribute__((aligned(32))) bf16_t PsF[4][32][16];
  int wid  = threadIdx.x >> 5;
  int lane = threadIdx.x & 31;
  int idx = blockIdx.x * 4 + wid;
  int qt = idx & 63;                 // S/16 = 64
  int h  = (idx >> 6) % HN;
  int b  = idx / (64 * HN);
  if (b >= BZ) return;               // wave-uniform; kernel has no barriers
  int q0 = qt * 16;
  int m = lane & 15, half = lane >> 4;
  const float scale = 0.125f;        // 1/sqrt(64)

  // Q fragments (16x64 -> two 16x32 A-frags) via 16B vector loads
  const bf16_t* Qr = Q + ((size_t)(b * SN + q0 + m) * DM + h * DHH);
  v8bf qa0 = *(const v8bf*)(Qr + half * 16);
  v8bf qa1 = *(const v8bf*)(Qr + 8 + half * 16);
  v8bf qa2 = *(const v8bf*)(Qr + 32 + half * 16);
  v8bf qa3 = *(const v8bf*)(Qr + 40 + half * 16);
  v16bf qf1 = __builtin_shufflevector(qa0, qa1, 0, 1, 2, 3, 4, 5, 6, 7, 8, 9,
                                      10, 11, 12, 13, 14, 15);
  v16bf qf2 = __builtin_shufflevector(qa2, qa3, 0, 1, 2, 3, 4, 5, 6, 7, 8, 9,
                                      10, 11, 12, 13, 14, 15);

  float mprev[8], lsum[8];
  v8f oacc[4] = {{}, {}, {}, {}};
#pragma unroll
  for (int r = 0; r < 8; ++r) { mprev[r] = -3.0e38f; lsum[r] = 0.f; }

  // per-lane transposed-V column base: d = g*16 + m (g handled by offset)
  const bf16_t* Vt0 = Vt + ((size_t)(b * HN + h) * DHH + m) * SN;
  const int plane = half * 8;  // + r at use
  const int pslot = m;

  auto qkS = [&](int kt, v8f* sacc) {
    const bf16_t* Kr = K + ((size_t)(b * SN + kt * 16 + m) * DM + h * DHH);
    v16bf kf1 = *(const v16bf*)(Kr + half * 16);
    v16bf kf2 = *(const v16bf*)(Kr + 32 + half * 16);
    *sacc = __builtin_amdgcn_wmma_f32_16x16x32_bf16(false, qf1, false, kf1,
                                                    (short)0, *sacc, false, false);
    *sacc = __builtin_amdgcn_wmma_f32_16x16x32_bf16(false, qf2, false, kf2,
                                                    (short)0, *sacc, false, false);
  };
  auto pv = [&](int soff) {  // B-frag: one 32B vector load per d-group
#pragma unroll
    for (int g = 0; g < 4; ++g) {
      v16bf vf = *(const v16bf*)(Vt0 + (size_t)g * 16 * SN + soff);
      oacc[g] = __builtin_amdgcn_wmma_f32_16x16x32_bf16(
          false, *(const v16bf*)&PsF[wid][lane][0], false, vf, (short)0,
          oacc[g], false, false);
    }
  };

  const int npairs = (qt + 1) >> 1;
  for (int pz = 0; pz < npairs; ++pz) {
    int kt = pz * 2;
    v8f s1 = {}, s2 = {};
    qkS(kt, &s1);
    qkS(kt + 1, &s2);
    const bool diag2 = (kt + 1 == qt);
#pragma unroll
    for (int r = 0; r < 8; ++r) {
      float a = s1[r] * scale;
      float c = s2[r] * scale;
      int qg = q0 + half * 8 + r;
      if (diag2 && (kt * 16 + 16 + m) > qg) c = -1.0e30f;
      float mx = fmaxf(a, c);
#pragma unroll
      for (int k2 = 8; k2; k2 >>= 1) mx = fmaxf(mx, __shfl_xor(mx, k2, 32));
      float mn = fmaxf(mprev[r], mx);
      float alpha = __expf(mprev[r] - mn);
      float p1 = __expf(a - mn);
      float p2 = __expf(c - mn);
      float rs = p1 + p2;
#pragma unroll
      for (int k2 = 8; k2; k2 >>= 1) rs += __shfl_xor(rs, k2, 32);
      lsum[r] = lsum[r] * alpha + rs;
      mprev[r] = mn;
#pragma unroll
      for (int g = 0; g < 4; ++g) oacc[g][r] *= alpha;
      PsF[wid][plane + r][pslot] = (bf16_t)p1;       // k-tile 1 -> lanes 0..15
      PsF[wid][plane + 16 + r][pslot] = (bf16_t)p2;  // k-tile 2 -> lanes 16..31
    }
    pv(kt * 16 + half * 16);   // full K=32: half0 -> s..s+15, half1 -> s+16..s+31
  }
  if ((qt + 1) & 1) {          // remainder (diagonal) tile, K padded with zeros
    int kt = qt;
    v8f s1 = {};
    qkS(kt, &s1);
#pragma unroll
    for (int r = 0; r < 8; ++r) {
      float a = s1[r] * scale;
      int qg = q0 + half * 8 + r;
      if ((kt * 16 + m) > qg) a = -1.0e30f;
      float mx = a;
#pragma unroll
      for (int k2 = 8; k2; k2 >>= 1) mx = fmaxf(mx, __shfl_xor(mx, k2, 32));
      float mn = fmaxf(mprev[r], mx);
      float alpha = __expf(mprev[r] - mn);
      float p1 = __expf(a - mn);
      float rs = p1;
#pragma unroll
      for (int k2 = 8; k2; k2 >>= 1) rs += __shfl_xor(rs, k2, 32);
      lsum[r] = lsum[r] * alpha + rs;
      mprev[r] = mn;
#pragma unroll
      for (int g = 0; g < 4; ++g) oacc[g][r] *= alpha;
      PsF[wid][plane + r][pslot] = (bf16_t)p1;
      PsF[wid][plane + 16 + r][pslot] = (bf16_t)0.f;  // zero the pad half
    }
    pv(kt * 16);               // both halves load same (valid) 16 rows
  }

#pragma unroll
  for (int r = 0; r < 8; ++r) {
    float inv = 1.f / lsum[r];
    int qg = q0 + half * 8 + r;
#pragma unroll
    for (int g = 0; g < 4; ++g)
      O[((size_t)(b * SN + qg) * DM) + h * DHH + g * 16 + m] =
          (bf16_t)(oacc[g][r] * inv);
  }
}

// ---------------------------------------------------------------------------
// Host launch sequence
// ---------------------------------------------------------------------------
extern "C" void kernel_launch(void* const* d_in, const int* in_sizes, int n_in,
                              void* d_out, int out_size, void* d_ws,
                              size_t ws_size, hipStream_t stream) {
  const int*   ids   = (const int*)d_in[0];
  const float* tok   = (const float*)d_in[1];
  const float* pos   = (const float*)d_in[2];
  const float* wq    = (const float*)d_in[3];
  const float* bq    = (const float*)d_in[4];
  const float* wk    = (const float*)d_in[5];
  const float* bk    = (const float*)d_in[6];
  const float* wv    = (const float*)d_in[7];
  const float* bv    = (const float*)d_in[8];
  const float* wo    = (const float*)d_in[9];
  const float* bo    = (const float*)d_in[10];
  const float* ln1w  = (const float*)d_in[11];
  const float* ln1b  = (const float*)d_in[12];
  const float* ln2w  = (const float*)d_in[13];
  const float* ln2b  = (const float*)d_in[14];
  const float* fc1w  = (const float*)d_in[15];
  const float* fc1b  = (const float*)d_in[16];
  const float* fc2w  = (const float*)d_in[17];
  const float* fc2b  = (const float*)d_in[18];
  const float* lnfw  = (const float*)d_in[19];
  const float* lnfb  = (const float*)d_in[20];

  char* ws = (char*)d_ws;
  size_t off = 0;
  auto carve = [&](size_t bytes) -> char* {
    char* p = ws + off;
    off = (off + bytes + 255) & ~(size_t)255;
    return p;
  };
  const size_t TOK = (size_t)BZ * SN;
  float*  x    = (float*)carve(TOK * DM * 4);
  bf16_t* hb   = (bf16_t*)carve(TOK * DM * 2);
  bf16_t* qb   = (bf16_t*)carve(TOK * DM * 2);
  bf16_t* kb   = (bf16_t*)carve(TOK * DM * 2);
  bf16_t* vtb  = (bf16_t*)carve(TOK * DM * 2);   // transposed V
  bf16_t* ab   = (bf16_t*)carve(TOK * DM * 2);
  bf16_t* midb = (bf16_t*)carve(TOK * DFFN * 2);
  bf16_t* xfb  = (bf16_t*)carve((size_t)BZ * DM * 2);

  int total = (int)(TOK * DM);
  gpt2_embed_kernel<<<(total + 255) / 256, 256, 0, stream>>>(ids, tok, pos, x,
                                                             total);

  dim3 gD(DM / 64, (int)(TOK / 128));      // N=768 GEMMs
  dim3 gF(DFFN / 64, (int)(TOK / 128));    // N=3072 GEMM
  int attnGrid = (BZ * HN * (SN / 16)) / 4;

  for (int l = 0; l < LN_; ++l) {
    const float* lwq = wq + (size_t)l * DM * DM;
    const float* lwk = wk + (size_t)l * DM * DM;
    const float* lwv = wv + (size_t)l * DM * DM;
    const float* lwo = wo + (size_t)l * DM * DM;
    const float* lf1 = fc1w + (size_t)l * DM * DFFN;
    const float* lf2 = fc2w + (size_t)l * DFFN * DM;

    gpt2_ln_kernel<<<(int)TOK, 256, 0, stream>>>(x, ln1w + l * DM, ln1b + l * DM,
                                                 hb, 1, 0);
    gpt2_gemm_kernel<<<gD, 256, 0, stream>>>(hb, lwq, bq + l * DM, nullptr,
                                             nullptr, qb, (int)TOK, DM, DM,
                                             0, 0, 0);
    gpt2_gemm_kernel<<<gD, 256, 0, stream>>>(hb, lwk, bk + l * DM, nullptr,
                                             nullptr, kb, (int)TOK, DM, DM,
                                             0, 0, 0);
    gpt2_gemm_kernel<<<gD, 256, 0, stream>>>(hb, lwv, bv + l * DM, nullptr,
                                             nullptr, vtb, (int)TOK, DM, DM,
                                             0, 0, 1);
    gpt2_attn_kernel<<<attnGrid, 128, 0, stream>>>(qb, kb, vtb, ab);
    gpt2_gemm_kernel<<<gD, 256, 0, stream>>>(ab, lwo, bo + l * DM, x, x,
                                             nullptr, (int)TOK, DM, DM, 0, 0, 0);
    gpt2_ln_kernel<<<(int)TOK, 256, 0, stream>>>(x, ln2w + l * DM, ln2b + l * DM,
                                                 hb, 1, 0);
    gpt2_gemm_kernel<<<gF, 256, 0, stream>>>(hb, lf1, fc1b + l * DFFN, nullptr,
                                             nullptr, midb, (int)TOK, DFFN, DM,
                                             0, 1, 0);
    gpt2_gemm_kernel<<<gD, 256, 0, stream>>>(midb, lf2, fc2b + l * DM, x, x,
                                             nullptr, (int)TOK, DM, DFFN,
                                             0, 0, 0);
  }

  // final LN on the 8 last-token rows, then tied lm_head (W^T access)
  gpt2_ln_kernel<<<BZ, 256, 0, stream>>>(x, lnfw, lnfb, xfb, SN, SN - 1);
  dim3 gV((VN + 63) / 64, 1);
  gpt2_gemm_kernel<<<gV, 256, 0, stream>>>(xfb, tok, nullptr, nullptr,
                                           (float*)d_out, nullptr, BZ, VN, DM,
                                           1, 0, 0);
}
